// MoEModel_18476949307967
// MI455X (gfx1250) — matrix operand order repaired and checked by reading, not compiled, not run on hardware
//
#include <hip/hip_runtime.h>
#include <hip/hip_bf16.h>

// ---------------------------------------------------------------------------
// Model dims (fixed by the reference)
// ---------------------------------------------------------------------------
#define BB   2
#define SS   2048
#define DD   1024
#define EE   8
#define LL   4
#define HH   8
#define HDIM 128
#define TT   (BB * SS)          // 4096 tokens

typedef __attribute__((ext_vector_type(16))) __bf16 v16bf;
typedef __attribute__((ext_vector_type(8)))  float  v8f;

union AFrag { uint4 q[2]; v16bf v; unsigned short u[16]; };
union BFrag { uint4 q[2]; v16bf v; unsigned short u[16]; };

__device__ __forceinline__ unsigned short f2bf(float f) {
  unsigned int u = __float_as_uint(f);
  u += 0x7FFFu + ((u >> 16) & 1u);          // round-to-nearest-even
  return (unsigned short)(u >> 16);
}

// ---------------------------------------------------------------------------
// WMMA GEMM:  C[M,N] = A(bf16,[M,K] rm) x W(bf16,[N,K] rm)  (+bias)(+relu)...
//   flags: 1=relu, 2=store fp32 Cf, 4=store bf16 Cb,
//          8=Cf += gate[row,gi]*val, 16=Cf = gate[row,gi]*val
// Block: 256 thr = 8 waves; wave -> 32(m) x 64(n). Block tile 256m x 64n.
// Inner loop per k-step(32): 4 A b128 loads + 8 B b128 loads + 8 WMMA.
// ---------------------------------------------------------------------------
__global__ __launch_bounds__(256)
void gemm_wmma(const unsigned short* __restrict__ A,
               const unsigned short* __restrict__ W,
               const float* __restrict__ bias,
               float* __restrict__ Cf,
               unsigned short* __restrict__ Cb,
               const float* __restrict__ gate, int gateStride, int gateIdx,
               int M, int N, int Kd, int flags)
{
  const int tid  = threadIdx.x;
  const int wave = tid >> 5;
  const int lane = tid & 31;
  const int l15  = lane & 15;
  const int hi   = lane >> 4;
  const int m0   = blockIdx.y * 256 + wave * 32;
  const int n0   = blockIdx.x * 64;

  v8f cc[2][4];
#pragma unroll
  for (int mi = 0; mi < 2; ++mi)
#pragma unroll
    for (int j = 0; j < 4; ++j) cc[mi][j] = {};

  const unsigned short* ar0 = A + (size_t)(m0 + l15) * Kd + 8 * hi;
  const unsigned short* ar1 = ar0 + (size_t)16 * Kd;
  const unsigned short* wrow[4];
#pragma unroll
  for (int j = 0; j < 4; ++j)
    wrow[j] = W + (size_t)(n0 + j * 16 + l15) * Kd + 8 * hi;

  for (int kk = 0; kk < Kd; kk += 32) {
    AFrag a0, a1;
    a0.q[0] = *(const uint4*)(ar0 + kk);
    a0.q[1] = *(const uint4*)(ar0 + kk + 16);
    a1.q[0] = *(const uint4*)(ar1 + kk);
    a1.q[1] = *(const uint4*)(ar1 + kk + 16);

    BFrag b[4];
#pragma unroll
    for (int j = 0; j < 4; ++j) {
      b[j].q[0] = *(const uint4*)(wrow[j] + kk);
      b[j].q[1] = *(const uint4*)(wrow[j] + kk + 16);
    }
#pragma unroll
    for (int j = 0; j < 4; ++j) {
      cc[0][j] = __builtin_amdgcn_wmma_f32_16x16x32_bf16(
          false, a0.v, false, b[j].v, (short)0, cc[0][j], false, false);
      cc[1][j] = __builtin_amdgcn_wmma_f32_16x16x32_bf16(
          false, a1.v, false, b[j].v, (short)0, cc[1][j], false, false);
    }
  }

#pragma unroll
  for (int mi = 0; mi < 2; ++mi) {
#pragma unroll
    for (int j = 0; j < 4; ++j) {
      const int n  = n0 + j * 16 + l15;
      const float bv = bias ? bias[n] : 0.0f;
#pragma unroll
      for (int v = 0; v < 8; ++v) {
        const int row = m0 + mi * 16 + v + 8 * hi;
        float val = cc[mi][j][v] + bv;
        if (flags & 1) val = fmaxf(val, 0.0f);
        const size_t idx = (size_t)row * N + n;
        if (flags & 2) Cf[idx] = val;
        if (flags & 4) Cb[idx] = f2bf(val);
        if (flags & 24) {
          const float g = gate[(size_t)row * gateStride + gateIdx];
          if (flags & 8)  Cf[idx] += g * val;
          else            Cf[idx]  = g * val;
        }
      }
    }
  }
}

// ---------------------------------------------------------------------------
// fp32 [K,N] -> bf16 [N,K] tiled transpose+convert (per expert slab)
// grid (N/32, K/32, E), block (32,8)
// ---------------------------------------------------------------------------
__global__ __launch_bounds__(256)
void transpose_cvt(const float* __restrict__ in, unsigned short* __restrict__ out)
{
  __shared__ unsigned short tile[32][33];
  const size_t base = (size_t)blockIdx.z * DD * DD;
  const int x = blockIdx.x * 32 + threadIdx.x;          // input col (n)
#pragma unroll
  for (int j = 0; j < 4; ++j) {
    const int y = blockIdx.y * 32 + threadIdx.y + j * 8; // input row (k)
    tile[threadIdx.y + j * 8][threadIdx.x] = f2bf(in[base + (size_t)y * DD + x]);
  }
  __syncthreads();
  const int x2 = blockIdx.y * 32 + threadIdx.x;          // output col (k)
#pragma unroll
  for (int j = 0; j < 4; ++j) {
    const int y2 = blockIdx.x * 32 + threadIdx.y + j * 8; // output row (n)
    out[base + (size_t)y2 * DD + x2] = tile[threadIdx.x][threadIdx.y + j * 8];
  }
}

// ---------------------------------------------------------------------------
// Router logits: one block per token, one wave per expert (fp32 exact path;
// logits are a validated output).
// ---------------------------------------------------------------------------
__global__ __launch_bounds__(256)
void router_kernel(const float* __restrict__ x, const float* __restrict__ rw,
                   const float* __restrict__ rb, float* __restrict__ logits)
{
  const int t = blockIdx.x;
  const int e = threadIdx.x >> 5;
  const int lane = threadIdx.x & 31;
  const float* xr = x + (size_t)t * DD;
  float s = 0.0f;
  for (int d = lane; d < DD; d += 32)
    s += xr[d] * rw[(size_t)d * EE + e];
#pragma unroll
  for (int m = 16; m; m >>= 1) s += __shfl_xor(s, m);
  if (lane == 0) logits[(size_t)t * EE + e] = s + rb[e];
}

// ---------------------------------------------------------------------------
// softmax over E=8 + top-2 -> gate weights (0 for non-selected experts)
// ---------------------------------------------------------------------------
__global__ void gate_kernel(const float* __restrict__ logits, float* __restrict__ gate)
{
  const int t = blockIdx.x * blockDim.x + threadIdx.x;
  if (t >= TT) return;
  float l[EE], p[EE];
  float mx = -3.0e38f;
#pragma unroll
  for (int e = 0; e < EE; ++e) { l[e] = logits[(size_t)t * EE + e]; mx = fmaxf(mx, l[e]); }
  float sum = 0.0f;
#pragma unroll
  for (int e = 0; e < EE; ++e) { p[e] = __expf(l[e] - mx); sum += p[e]; }
  const float inv = 1.0f / sum;
#pragma unroll
  for (int e = 0; e < EE; ++e) p[e] *= inv;
  int i0 = 0;
#pragma unroll
  for (int e = 1; e < EE; ++e) if (p[e] > p[i0]) i0 = e;
  int i1 = -1;
#pragma unroll
  for (int e = 0; e < EE; ++e)
    if (e != i0 && (i1 < 0 || p[e] > p[i1])) i1 = e;
#pragma unroll
  for (int e = 0; e < EE; ++e)
    gate[(size_t)t * EE + e] = (e == i0 || e == i1) ? p[e] : 0.0f;
}

__global__ void cvt_kernel(const float* __restrict__ in, unsigned short* __restrict__ out, int n) {
  const int i = blockIdx.x * blockDim.x + threadIdx.x;
  if (i < n) out[i] = f2bf(in[i]);
}

// ---------------------------------------------------------------------------
// x = LayerNorm(res + xin) * g + b ; write fp32 x, bf16 x, and res=x
// ---------------------------------------------------------------------------
__global__ __launch_bounds__(256)
void addln_kernel(const float* __restrict__ resid, const float* __restrict__ xin,
                  const float* __restrict__ g, const float* __restrict__ bta,
                  float* __restrict__ xo, unsigned short* __restrict__ xob,
                  float* __restrict__ resout)
{
  const int t = blockIdx.x;
  __shared__ float red[256];
  float vals[4];
  const float* r  = resid + (size_t)t * DD;
  const float* xi = xin   + (size_t)t * DD;
  float s = 0.0f;
#pragma unroll
  for (int j = 0; j < 4; ++j) {
    const int d = threadIdx.x + j * 256;
    vals[j] = r[d] + xi[d];
    s += vals[j];
  }
  red[threadIdx.x] = s; __syncthreads();
  for (int st = 128; st; st >>= 1) { if (threadIdx.x < st) red[threadIdx.x] += red[threadIdx.x + st]; __syncthreads(); }
  const float mean = red[0] / DD;
  __syncthreads();
  float s2 = 0.0f;
#pragma unroll
  for (int j = 0; j < 4; ++j) { const float dv = vals[j] - mean; s2 += dv * dv; }
  red[threadIdx.x] = s2; __syncthreads();
  for (int st = 128; st; st >>= 1) { if (threadIdx.x < st) red[threadIdx.x] += red[threadIdx.x + st]; __syncthreads(); }
  const float rstd = rsqrtf(red[0] / DD + 1e-5f);
#pragma unroll
  for (int j = 0; j < 4; ++j) {
    const int d = threadIdx.x + j * 256;
    const float o = (vals[j] - mean) * rstd * g[d] + bta[d];
    const size_t idx = (size_t)t * DD + d;
    xo[idx] = o; xob[idx] = f2bf(o); resout[idx] = o;
  }
}

// ---------------------------------------------------------------------------
// Flash attention (non-causal), bf16 WMMA, fp32 accumulators.
// qkv: [T, 3D] bf16 (q | k | v), head h at column h*HDIM.
// Grid: (SS/64, BB*HH), block 128 (4 waves); each wave: 16 q-rows x HD=128.
// ---------------------------------------------------------------------------
__global__ __launch_bounds__(128)
void flash_kernel(const unsigned short* __restrict__ qkv, unsigned short* __restrict__ o)
{
  const int wave = threadIdx.x >> 5;
  const int lane = threadIdx.x & 31;
  const int l15  = lane & 15;
  const int hi   = lane >> 4;
  const int bh = blockIdx.y, b = bh >> 3, h = bh & 7;
  const int m0 = blockIdx.x * 64 + wave * 16;
  const size_t rowStride = 3 * DD;
  const size_t baseT = (size_t)b * SS;

  __shared__ __align__(16) unsigned short vls[4][32 * HDIM]; // 32 KB
  __shared__ __align__(16) unsigned short pls[4][16 * 32];   // 4 KB
  unsigned short* vsh = vls[wave];
  unsigned short* psh = pls[wave];

  AFrag qa[4];
  {
    const unsigned short* qrow =
        qkv + (baseT + m0 + l15) * rowStride + (size_t)h * HDIM + 8 * hi;
#pragma unroll
    for (int dk = 0; dk < 4; ++dk) {
      qa[dk].q[0] = *(const uint4*)(qrow + dk * 32);
      qa[dk].q[1] = *(const uint4*)(qrow + dk * 32 + 16);
    }
  }

  v8f acc[8];
#pragma unroll
  for (int n = 0; n < 8; ++n) acc[n] = {};
  float rmax[8], rsum[8];
#pragma unroll
  for (int v = 0; v < 8; ++v) { rmax[v] = -3.0e38f; rsum[v] = 0.0f; }
  const float scale = 0.08838834764831845f; // 1/sqrt(128)

  for (int kc = 0; kc < SS; kc += 32) {
    {
      const unsigned short* vrow =
          qkv + (baseT + kc + lane) * rowStride + 2 * DD + (size_t)h * HDIM;
      unsigned short* dst = vsh + lane * HDIM;
#pragma unroll
      for (int q4 = 0; q4 < 16; ++q4)
        *(uint4*)(dst + q4 * 8) = *(const uint4*)(vrow + q4 * 8);
    }

    v8f s0 = {}, s1 = {};
#pragma unroll
    for (int dk = 0; dk < 4; ++dk) {
      BFrag kb0, kb1;
      const unsigned short* k0r =
          qkv + (baseT + kc + l15) * rowStride + DD + (size_t)h * HDIM + dk * 32 + 8 * hi;
      const unsigned short* k1r = k0r + 16 * rowStride;
      kb0.q[0] = *(const uint4*)(k0r);
      kb0.q[1] = *(const uint4*)(k0r + 16);
      kb1.q[0] = *(const uint4*)(k1r);
      kb1.q[1] = *(const uint4*)(k1r + 16);
      s0 = __builtin_amdgcn_wmma_f32_16x16x32_bf16(false, qa[dk].v, false, kb0.v, (short)0, s0, false, false);
      s1 = __builtin_amdgcn_wmma_f32_16x16x32_bf16(false, qa[dk].v, false, kb1.v, (short)0, s1, false, false);
    }

    float corr[8];
#pragma unroll
    for (int v = 0; v < 8; ++v) {
      const float a0 = s0[v] * scale, a1 = s1[v] * scale;
      float mx = fmaxf(a0, a1);
      mx = fmaxf(mx, __shfl_xor(mx, 1));
      mx = fmaxf(mx, __shfl_xor(mx, 2));
      mx = fmaxf(mx, __shfl_xor(mx, 4));
      mx = fmaxf(mx, __shfl_xor(mx, 8));
      const float nm = fmaxf(rmax[v], mx);
      corr[v] = __expf(rmax[v] - nm);
      rmax[v] = nm;
      const float p0 = __expf(a0 - nm), p1 = __expf(a1 - nm);
      s0[v] = p0; s1[v] = p1;
      float ps = p0 + p1;
      ps += __shfl_xor(ps, 1); ps += __shfl_xor(ps, 2);
      ps += __shfl_xor(ps, 4); ps += __shfl_xor(ps, 8);
      rsum[v] = rsum[v] * corr[v] + ps;
    }
#pragma unroll
    for (int n = 0; n < 8; ++n)
#pragma unroll
      for (int v = 0; v < 8; ++v) acc[n][v] *= corr[v];

#pragma unroll
    for (int v = 0; v < 8; ++v) {
      const int row = v + 8 * hi;
      psh[row * 32 + l15]      = f2bf(s0[v]);
      psh[row * 32 + 16 + l15] = f2bf(s1[v]);
    }
    asm volatile("s_wait_dscnt 0" ::: "memory");
    AFrag pa;
    {
      const unsigned short* pr = psh + l15 * 32 + 8 * hi;
      pa.q[0] = *(const uint4*)pr;
      pa.q[1] = *(const uint4*)(pr + 16);
    }

#pragma unroll
    for (int n = 0; n < 8; ++n) {
      BFrag vb;
      const unsigned short* vcol = vsh + (8 * hi) * HDIM + n * 16 + l15;
#pragma unroll
      for (int e = 0; e < 16; ++e) vb.u[e] = vcol[(e + (e & 8)) * HDIM];
      acc[n] = __builtin_amdgcn_wmma_f32_16x16x32_bf16(false, pa.v, false, vb.v, (short)0, acc[n], false, false);
    }
  }

#pragma unroll
  for (int v = 0; v < 8; ++v) {
    const float inv = 1.0f / rsum[v];
    const int row = m0 + v + 8 * hi;
    unsigned short* orow = o + (baseT + row) * (size_t)DD + (size_t)h * HDIM;
#pragma unroll
    for (int n = 0; n < 8; ++n)
      orow[n * 16 + l15] = f2bf(acc[n][v] * inv);
  }
}

// ---------------------------------------------------------------------------
// Host orchestration
// ---------------------------------------------------------------------------
extern "C" void kernel_launch(void* const* d_in, const int* in_sizes, int n_in,
                              void* d_out, int out_size, void* d_ws, size_t ws_size,
                              hipStream_t stream)
{
  (void)in_sizes; (void)n_in; (void)out_size; (void)ws_size;

  const float* x_in  = (const float*)d_in[0];
  const float* inp_w = (const float*)d_in[1];
  const float* inp_b = (const float*)d_in[2];
  const float* ln_g  = (const float*)d_in[3];
  const float* ln_b  = (const float*)d_in[4];

  struct Lyr { const float *rw, *rb, *w1, *b1, *w2, *b2; } lyr[LL];
  for (int i = 0; i < LL; ++i) {
    const int base = 5 + 6 * i;
    lyr[i].rw = (const float*)d_in[base + 0];
    lyr[i].rb = (const float*)d_in[base + 1];
    lyr[i].w1 = (const float*)d_in[base + 2];
    lyr[i].b1 = (const float*)d_in[base + 3];
    lyr[i].w2 = (const float*)d_in[base + 4];
    lyr[i].b2 = (const float*)d_in[base + 5];
  }
  struct Att { const float *qw, *qb, *ow, *ob; } att[LL - 1];
  for (int j = 0; j < LL - 1; ++j) {
    const int base = 5 + 6 * LL + 4 * j;   // = 29 + 4*j
    att[j].qw = (const float*)d_in[base + 0];
    att[j].qb = (const float*)d_in[base + 1];
    att[j].ow = (const float*)d_in[base + 2];
    att[j].ob = (const float*)d_in[base + 3];
  }

  const size_t TD = (size_t)TT * DD;   // 4M
  const size_t WE = (size_t)EE * DD * DD; // 8M (one layer's w1 or w2)
  float* xbuf  = (float*)d_ws;
  float* res   = xbuf + TD;
  float* xatt  = res  + TD;
  float* moe   = xatt + TD;
  float* gateb = moe  + TD;
  unsigned short* xb    = (unsigned short*)(gateb + (size_t)TT * EE);
  unsigned short* hb    = xb + TD;
  unsigned short* qkvb  = hb + TD;
  unsigned short* attnb = qkvb + 3 * TD;
  unsigned short* wt1   = attnb + TD;        // bf16 [E,N,K] for current layer's w1
  unsigned short* wt2   = wt1 + WE;          // bf16 [E,N,K] for current layer's w2
  unsigned short* wnk   = wt2 + WE;          // bf16 NK staging (up to 3*D*D)

  float* out_x = (float*)d_out;
  const int nTD = TT * DD;

  auto GEMM = [&](const unsigned short* A, const unsigned short* W, const float* bias,
                  float* Cf, unsigned short* Cb, const float* gate, int gi,
                  int M, int N, int flags) {
    dim3 grid(N / 64, M / 256);
    gemm_wmma<<<grid, 256, 0, stream>>>(A, W, bias, Cf, Cb, gate, EE, gi,
                                        M, N, DD, flags);
  };

  // input projection: x = x_in @ inp_w^T + inp_b   (write fp32 + bf16)
  cvt_kernel<<<nTD / 256, 256, 0, stream>>>(x_in, hb, nTD);
  cvt_kernel<<<(DD * DD) / 256, 256, 0, stream>>>(inp_w, wnk, DD * DD);
  GEMM(hb, wnk, inp_b, xbuf, xb, nullptr, 0, TT, DD, 2 | 4);
  hipMemcpyAsync(res, xbuf, TD * sizeof(float), hipMemcpyDeviceToDevice, stream);

  for (int i = 0; i < LL; ++i) {
    float* logits = out_x + TD + (size_t)i * TT * EE;   // routing-logits output slot
    router_kernel<<<TT, 256, 0, stream>>>(xbuf, lyr[i].rw, lyr[i].rb, logits);
    gate_kernel<<<TT / 256, 256, 0, stream>>>(logits, gateb);

    // bf16-NK copies of this layer's expert weights
    transpose_cvt<<<dim3(DD / 32, DD / 32, EE), dim3(32, 8), 0, stream>>>(lyr[i].w1, wt1);
    transpose_cvt<<<dim3(DD / 32, DD / 32, EE), dim3(32, 8), 0, stream>>>(lyr[i].w2, wt2);

    float* accp = (i < LL - 1) ? moe : out_x;           // last layer writes d_out
    for (int e = 0; e < EE; ++e) {
      // h = relu(x @ w1[e] + b1[e])  (bf16 out)
      GEMM(xb, wt1 + (size_t)e * DD * DD, lyr[i].b1 + (size_t)e * DD,
           nullptr, hb, nullptr, 0, TT, DD, 1 | 4);
      // acc (=|+=) gate[:,e] * (h @ w2[e] + b2[e])
      GEMM(hb, wt2 + (size_t)e * DD * DD, lyr[i].b2 + (size_t)e * DD,
           accp, nullptr, gateb, e, TT, DD, (e == 0) ? 16 : 8);
    }

    if (i < LL - 1) {
      cvt_kernel<<<nTD / 256, 256, 0, stream>>>(moe, xb, nTD);
      // qkv projection (N = 3D), bf16 out
      cvt_kernel<<<(3 * DD * DD) / 256, 256, 0, stream>>>(att[i].qw, wnk, 3 * DD * DD);
      GEMM(xb, wnk, att[i].qb, nullptr, qkvb, nullptr, 0, TT, 3 * DD, 4);
      flash_kernel<<<dim3(SS / 64, BB * HH), 128, 0, stream>>>(qkvb, attnb);
      // output projection, fp32 out
      cvt_kernel<<<(DD * DD) / 256, 256, 0, stream>>>(att[i].ow, wnk, DD * DD);
      GEMM(attnb, wnk, att[i].ob, xatt, nullptr, nullptr, 0, TT, DD, 2);
      // x = LN(res + xatt); refresh bf16 copy and next residual
      addln_kernel<<<TT, 256, 0, stream>>>(res, xatt, ln_g, ln_b, xbuf, xb, res);
    }
  }
}
